// Mamba_76055280877998
// MI455X (gfx1250) — compile-verified, matrix operand order checked by reading, not run on hardware
//
#include <hip/hip_runtime.h>
#include <hip/hip_bf16.h>

// ---------------- model dims (match reference) ----------------
#define N_LAYERS 4
#define D_MODEL  1024
#define D_INNER  2048            // EXPAND * D_MODEL
#define D_STATE  16
#define DT_RANK  64
#define K_CONV   4
#define OUT_DIM  1024
#define BATCH    2
#define SEQ      1024
#define MTOT     (BATCH * SEQ)   // 2048 rows for every GEMM
#define DBC_N    (DT_RANK + 2 * D_STATE)  // 96

typedef __attribute__((ext_vector_type(16))) __bf16 v16bf;
typedef __attribute__((ext_vector_type(8)))  float  v8f;

// ---------------- helpers ----------------
__device__ __forceinline__ __bf16 f2bf(float f) {
    union { float f; unsigned u; } a; a.f = f;
    unsigned r = a.u + 0x7FFFu + ((a.u >> 16) & 1u);   // round-to-nearest-even
    union { unsigned short s; __bf16 b; } o; o.s = (unsigned short)(r >> 16);
    return o.b;
}
__device__ __forceinline__ float sigmoid_f(float x) { return 1.f / (1.f + __expf(-x)); }
__device__ __forceinline__ float silu_f(float x)    { return x * sigmoid_f(x); }
__device__ __forceinline__ float softplus_f(float x){ return (x > 20.f) ? x : log1pf(__expf(x)); }

// ---------------- elementwise utility kernels ----------------
__global__ void copy_f32(const float* __restrict__ s, float* __restrict__ d, int n) {
    int i = blockIdx.x * blockDim.x + threadIdx.x;
    if (i < n) d[i] = s[i];
}

__global__ void cvt_bf16(const float* __restrict__ s, __bf16* __restrict__ d, int n) {
    int i = blockIdx.x * blockDim.x + threadIdx.x;
    if (i < n) d[i] = f2bf(s[i]);
}

// Wt[n*K + k] = bf16(W[k*N + n])  -- transpose so WMMA B fragments load contiguously
__global__ void wt_convert(const float* __restrict__ W, __bf16* __restrict__ Wt, int K, int N) {
    int i = blockIdx.x * blockDim.x + threadIdx.x;
    if (i >= K * N) return;
    int n = i / K, k = i - n * K;
    Wt[i] = f2bf(W[(size_t)k * N + n]);
}

// ---------------- RMSNorm: one block per row of D_MODEL ----------------
__global__ void rmsnorm_kernel(const float* __restrict__ x, const float* __restrict__ w,
                               __bf16* __restrict__ out) {
    __shared__ float red[256];
    __shared__ float s_rs;
    int row = blockIdx.x;
    const float* xr = x + (size_t)row * D_MODEL;
    float ss = 0.f;
    for (int i = threadIdx.x; i < D_MODEL; i += 256) { float v = xr[i]; ss += v * v; }
    red[threadIdx.x] = ss; __syncthreads();
    for (int s = 128; s > 0; s >>= 1) {
        if (threadIdx.x < s) red[threadIdx.x] += red[threadIdx.x + s];
        __syncthreads();
    }
    if (threadIdx.x == 0) s_rs = rsqrtf(red[0] * (1.f / D_MODEL) + 1e-5f);
    __syncthreads();
    float rs = s_rs;
    for (int i = threadIdx.x; i < D_MODEL; i += 256)
        out[(size_t)row * D_MODEL + i] = f2bf(xr[i] * rs * w[i]);
}

// ---------------- WMMA bf16 GEMM:  C[M,N] = A[M,K] @ B^T[N,K]  ----------------
// A row-major [M,K] bf16; B pre-transposed row-major [N,K] bf16; C float.
// Block = 256 threads = 8 waves, arranged 2(M) x 4(N); each wave computes a
// 32x32 macro-tile as 2x2 WMMA tiles (4 independent accumulators per K-step,
// 4 WMMAs per 4 fragments loaded -> 2x the WMMA/byte of a 1-tile wave).
// EP: 0 = none, 1 = +bias, 2 = softplus(acc+bias), 3 = +resid (residual add)
template<int EP>
__global__ void gemm_wmma(const __bf16* __restrict__ A, const __bf16* __restrict__ B,
                          const float* __restrict__ bias, const float* __restrict__ resid,
                          float* __restrict__ C, int M, int N, int K) {
    int lane = threadIdx.x & 31;
    int wave = threadIdx.x >> 5;
    int wm = wave >> 2;                        // 0..1
    int wn = wave & 3;                         // 0..3
    int m0 = blockIdx.x * 64 + wm * 32;
    int n0 = blockIdx.y * 128 + wn * 32;
    if (n0 >= N) return;                       // uniform per wave; EXEC stays all-1s

    int r  = lane & 15;                        // row (A) / col (B) within tile
    int hi = lane >> 4;                        // lanes 16-31 shifted by 8 in K
    const __bf16* ap0 = A + (size_t)(m0 + r) * K + hi * 8;
    const __bf16* ap1 = ap0 + (size_t)16 * K;
    const __bf16* bp0 = B + (size_t)(n0 + r) * K + hi * 8;
    const __bf16* bp1 = bp0 + (size_t)16 * K;

    v8f acc00 = {0.f,0.f,0.f,0.f,0.f,0.f,0.f,0.f};
    v8f acc01 = acc00, acc10 = acc00, acc11 = acc00;

    for (int kb = 0; kb < K; kb += 32) {
        // 16-bit fragment layout: elems 0..7 <- K = base+0..7, elems 8..15 <- K = base+16..23
        v16bf a0, a1, b0, b1;
#pragma unroll
        for (int i = 0; i < 8; i++) {
            a0[i] = ap0[i];  a0[8 + i] = ap0[16 + i];
            a1[i] = ap1[i];  a1[8 + i] = ap1[16 + i];
            b0[i] = bp0[i];  b0[8 + i] = bp0[16 + i];
            b1[i] = bp1[i];  b1[8 + i] = bp1[16 + i];
        }
        if (kb + 32 < K) {                      // WGP-scope prefetch of next K-tile
            __builtin_prefetch((const void*)(ap0 + 32), 0, 3);
            __builtin_prefetch((const void*)(ap1 + 32), 0, 3);
            __builtin_prefetch((const void*)(bp0 + 32), 0, 3);
            __builtin_prefetch((const void*)(bp1 + 32), 0, 3);
        }
        acc00 = __builtin_amdgcn_wmma_f32_16x16x32_bf16(false, a0, false, b0, (short)0, acc00, false, false);
        acc01 = __builtin_amdgcn_wmma_f32_16x16x32_bf16(false, a0, false, b1, (short)0, acc01, false, false);
        acc10 = __builtin_amdgcn_wmma_f32_16x16x32_bf16(false, a1, false, b0, (short)0, acc10, false, false);
        acc11 = __builtin_amdgcn_wmma_f32_16x16x32_bf16(false, a1, false, b1, (short)0, acc11, false, false);
        ap0 += 32; ap1 += 32; bp0 += 32; bp1 += 32;
    }

    // D layout: VGPR i holds M = mtile + i (+8 for lanes 16-31), N = ntile + (lane&15)
#pragma unroll
    for (int tm = 0; tm < 2; tm++) {
#pragma unroll
        for (int tn = 0; tn < 2; tn++) {
            const v8f& acc = tm ? (tn ? acc11 : acc10) : (tn ? acc01 : acc00);
            int col = n0 + tn * 16 + r;
            int mb  = m0 + tm * 16 + hi * 8;
#pragma unroll
            for (int i = 0; i < 8; i++) {
                float v = acc[i];
                size_t idx = (size_t)(mb + i) * N + col;
                if      (EP == 1) v += bias[col];
                else if (EP == 2) v  = softplus_f(v + bias[col]);
                else if (EP == 3) v += resid[idx];
                C[idx] = v;
            }
        }
    }
}

// ---------------- depthwise causal conv (K=4) + SiLU ----------------
// xz is [M, 2*D_INNER]; xp = xz[:, :D_INNER]. Writes xc (f32) and xc_bf (bf16 GEMM operand).
__global__ void conv_silu_kernel(const float* __restrict__ xz, const float* __restrict__ cw,
                                 const float* __restrict__ cb, float* __restrict__ xc,
                                 __bf16* __restrict__ xc_bf) {
    int g = blockIdx.x * blockDim.x + threadIdx.x;
    if (g >= MTOT * D_INNER) return;
    int m = g / D_INNER;
    int d = g - m * D_INNER;
    int t = m % SEQ;
    float acc = cb[d];
#pragma unroll
    for (int j = 0; j < K_CONV; j++) {
        int tt = t - (K_CONV - 1) + j;
        if (tt >= 0)
            acc += cw[d * K_CONV + j] * xz[(size_t)(m - t + tt) * (2 * D_INNER) + d];
    }
    float v = silu_f(acc);
    xc[g]    = v;
    xc_bf[g] = f2bf(v);
}

// dt = bf16(dbc[:, :DT_RANK])
__global__ void extract_dt(const float* __restrict__ dbc, __bf16* __restrict__ dt_bf) {
    int g = blockIdx.x * blockDim.x + threadIdx.x;
    if (g >= MTOT * DT_RANK) return;
    int m = g / DT_RANK, j = g - m * DT_RANK;
    dt_bf[g] = f2bf(dbc[(size_t)m * DBC_N + j]);
}

// ---------------- selective scan + gate ----------------
// One thread per (b, d) channel, h[16] in registers, sequential over SEQ.
// Bm/Cm (shared per timestep across the batch row) broadcast through LDS.
__global__ void scan_kernel(const float* __restrict__ u, const float* __restrict__ delta,
                            const float* __restrict__ dbc, const float* __restrict__ A_log,
                            const float* __restrict__ Dp, const float* __restrict__ xz,
                            __bf16* __restrict__ y_bf) {
    __shared__ float sB[D_STATE], sC[D_STATE];
    int b = blockIdx.x;
    int d = blockIdx.y * 256 + threadIdx.x;

    float Arow[D_STATE], h[D_STATE];
#pragma unroll
    for (int n = 0; n < D_STATE; n++) {
        Arow[n] = -__expf(A_log[(size_t)d * D_STATE + n]);
        h[n] = 0.f;
    }
    float Dd = Dp[d];

    for (int t = 0; t < SEQ; t++) {
        size_t m = (size_t)b * SEQ + t;
        if (threadIdx.x < D_STATE)
            sB[threadIdx.x] = dbc[m * DBC_N + DT_RANK + threadIdx.x];
        else if (threadIdx.x < 2 * D_STATE)
            sC[threadIdx.x - D_STATE] = dbc[m * DBC_N + DT_RANK + D_STATE + (threadIdx.x - D_STATE)];
        __syncthreads();

        float dl = delta[m * D_INNER + d];
        float uu = u[m * D_INNER + d];
        float du = dl * uu;
        float y = 0.f;
#pragma unroll
        for (int n = 0; n < D_STATE; n++) {
            h[n] = __expf(dl * Arow[n]) * h[n] + du * sB[n];
            y += h[n] * sC[n];
        }
        y += uu * Dd;
        float z = xz[m * (2 * D_INNER) + D_INNER + d];
        y *= silu_f(z);
        y_bf[m * D_INNER + d] = f2bf(y);
        __syncthreads();
    }
}

// ---------------- host orchestration ----------------
extern "C" void kernel_launch(void* const* d_in, const int* in_sizes, int n_in,
                              void* d_out, int out_size, void* d_ws, size_t ws_size,
                              hipStream_t stream) {
    const float* x_in   = (const float*)d_in[0];
    const float* norm_w = (const float*)d_in[1];
    const float* Win    = (const float*)d_in[2];
    const float* conv_w = (const float*)d_in[3];
    const float* conv_b = (const float*)d_in[4];
    const float* Wx     = (const float*)d_in[5];
    const float* Wdt    = (const float*)d_in[6];
    const float* bdt    = (const float*)d_in[7];
    const float* A_log  = (const float*)d_in[8];
    const float* Dp     = (const float*)d_in[9];
    const float* Wblk   = (const float*)d_in[10];
    const float* Wout   = (const float*)d_in[11];
    const float* bout   = (const float*)d_in[12];
    float* out = (float*)d_out;

    char* base = (char*)d_ws;
    size_t off = 0;
    auto wsalloc = [&](size_t bytes) -> void* {
        void* p = base + off;
        off += (bytes + 255) & ~(size_t)255;
        return p;
    };

    // bf16 transposed weights
    __bf16* win_bf  = (__bf16*)wsalloc((size_t)N_LAYERS * (2 * D_INNER) * D_MODEL * 2);
    __bf16* wx_bf   = (__bf16*)wsalloc((size_t)N_LAYERS * DBC_N * D_INNER * 2);
    __bf16* wdt_bf  = (__bf16*)wsalloc((size_t)N_LAYERS * D_INNER * DT_RANK * 2);
    __bf16* wblk_bf = (__bf16*)wsalloc((size_t)N_LAYERS * D_MODEL * D_INNER * 2);
    __bf16* wout_bf = (__bf16*)wsalloc((size_t)OUT_DIM * D_MODEL * 2);
    // activations
    float*  xcur  = (float*)wsalloc((size_t)MTOT * D_MODEL * 4);
    __bf16* xn_bf = (__bf16*)wsalloc((size_t)MTOT * D_MODEL * 2);
    float*  xz    = (float*)wsalloc((size_t)MTOT * 2 * D_INNER * 4);
    float*  xc    = (float*)wsalloc((size_t)MTOT * D_INNER * 4);
    __bf16* xc_bf = (__bf16*)wsalloc((size_t)MTOT * D_INNER * 2);
    float*  dbc   = (float*)wsalloc((size_t)MTOT * DBC_N * 4);
    __bf16* dt_bf = (__bf16*)wsalloc((size_t)MTOT * DT_RANK * 2);
    float*  delta = (float*)wsalloc((size_t)MTOT * D_INNER * 4);
    __bf16* y_bf  = (__bf16*)wsalloc((size_t)MTOT * D_INNER * 2);

    // one-time (per launch) weight transpose + bf16 conversion
    for (int l = 0; l < N_LAYERS; l++) {
        wt_convert<<<((size_t)D_MODEL * 2 * D_INNER + 255) / 256, 256, 0, stream>>>(
            Win + (size_t)l * D_MODEL * 2 * D_INNER,
            win_bf + (size_t)l * 2 * D_INNER * D_MODEL, D_MODEL, 2 * D_INNER);
        wt_convert<<<((size_t)D_INNER * DBC_N + 255) / 256, 256, 0, stream>>>(
            Wx + (size_t)l * D_INNER * DBC_N,
            wx_bf + (size_t)l * DBC_N * D_INNER, D_INNER, DBC_N);
        wt_convert<<<((size_t)DT_RANK * D_INNER + 255) / 256, 256, 0, stream>>>(
            Wdt + (size_t)l * DT_RANK * D_INNER,
            wdt_bf + (size_t)l * D_INNER * DT_RANK, DT_RANK, D_INNER);
        wt_convert<<<((size_t)D_INNER * D_MODEL + 255) / 256, 256, 0, stream>>>(
            Wblk + (size_t)l * D_INNER * D_MODEL,
            wblk_bf + (size_t)l * D_MODEL * D_INNER, D_INNER, D_MODEL);
    }
    wt_convert<<<((size_t)D_MODEL * OUT_DIM + 255) / 256, 256, 0, stream>>>(
        Wout, wout_bf, D_MODEL, OUT_DIM);

    copy_f32<<<(MTOT * D_MODEL + 255) / 256, 256, 0, stream>>>(x_in, xcur, MTOT * D_MODEL);

    for (int l = 0; l < N_LAYERS; l++) {
        const __bf16* win_l  = win_bf  + (size_t)l * 2 * D_INNER * D_MODEL;
        const __bf16* wx_l   = wx_bf   + (size_t)l * DBC_N * D_INNER;
        const __bf16* wdt_l  = wdt_bf  + (size_t)l * D_INNER * DT_RANK;
        const __bf16* wblk_l = wblk_bf + (size_t)l * D_MODEL * D_INNER;

        rmsnorm_kernel<<<MTOT, 256, 0, stream>>>(xcur, norm_w + (size_t)l * D_MODEL, xn_bf);

        // xz = xn @ Win      [2048 x 4096 x K=1024]
        gemm_wmma<0><<<dim3(MTOT / 64, (2 * D_INNER) / 128), 256, 0, stream>>>(
            xn_bf, win_l, nullptr, nullptr, xz, MTOT, 2 * D_INNER, D_MODEL);

        conv_silu_kernel<<<(MTOT * D_INNER) / 256, 256, 0, stream>>>(
            xz, conv_w + (size_t)l * D_INNER * K_CONV, conv_b + (size_t)l * D_INNER, xc, xc_bf);

        // dbc = xc @ Wx      [2048 x 96 x K=2048]
        gemm_wmma<0><<<dim3(MTOT / 64, 1), 256, 0, stream>>>(
            xc_bf, wx_l, nullptr, nullptr, dbc, MTOT, DBC_N, D_INNER);

        extract_dt<<<(MTOT * DT_RANK) / 256, 256, 0, stream>>>(dbc, dt_bf);

        // delta = softplus(dt @ Wdt + bdt)   [2048 x 2048 x K=64]
        gemm_wmma<2><<<dim3(MTOT / 64, D_INNER / 128), 256, 0, stream>>>(
            dt_bf, wdt_l, bdt + (size_t)l * D_INNER, nullptr, delta, MTOT, D_INNER, DT_RANK);

        scan_kernel<<<dim3(BATCH, D_INNER / 256), 256, 0, stream>>>(
            xc, delta, dbc, A_log + (size_t)l * D_INNER * D_STATE,
            Dp + (size_t)l * D_INNER, xz, y_bf);

        // x = x + y @ Wblk   [2048 x 1024 x K=2048], residual add fused
        gemm_wmma<3><<<dim3(MTOT / 64, D_MODEL / 128), 256, 0, stream>>>(
            y_bf, wblk_l, nullptr, xcur, xcur, MTOT, D_MODEL, D_INNER);
    }

    // out = x @ Wout + bout  [2048 x 1024 x K=1024]
    cvt_bf16<<<(MTOT * D_MODEL + 255) / 256, 256, 0, stream>>>(xcur, xn_bf, MTOT * D_MODEL);
    gemm_wmma<1><<<dim3(MTOT / 64, OUT_DIM / 128), 256, 0, stream>>>(
        xn_bf, wout_bf, bout, nullptr, out, MTOT, OUT_DIM, D_MODEL);
}
